// EGNNLayer_17806934409456
// MI455X (gfx1250) — matrix operand order repaired and checked by reading, not compiled, not run on hardware
//
#include <hip/hip_runtime.h>

// EGNN layer for MI455X (gfx1250, wave32, WMMA bf16).
// B=4, V=256, H=128.  Dominant GEMM: Ce = e @ Cw^T (262144x128x128) via
// v_wmma_f32_16x16x32_bf16.  Two passes over the 134MB edge tensor (global BN
// stats); pass 2 recomputes the GEMM instead of spilling e_new.
// Round 2: Cw B-fragments staged into LDS via global_load_async_to_lds_b128
// (ASYNCcnt) so the inner WMMA loop feeds from ds_load_b128 instead of a
// serialized global-load/wait chain.

#define NB 4
#define NV 256
#define NH 128
#define EPS 1e-5f
#define M_E (NB*NV*NV)          // 262144 edge rows

typedef __attribute__((ext_vector_type(16))) __bf16 v16bf;
typedef __attribute__((ext_vector_type(8)))  float  v8f;

#define WMMA_BF16(a,b,c) \
  __builtin_amdgcn_wmma_f32_16x16x32_bf16(false,(a),false,(b),(short)0,(c),false,false)

// generic LDS pointer -> hardware LDS byte address (low 32 bits hold it)
__device__ __forceinline__ unsigned lds_addr_u32(const void* p) {
  return (unsigned)(unsigned long long)p;
}

// Stage 32KB of B-fragments global->LDS with the CDNA5 async copy engine.
// 2048 x b128 transfers spread over 128 threads (16 each).
__device__ __forceinline__ void stage_frags_async(const __bf16* __restrict__ g,
                                                  __bf16* s, int tid) {
  unsigned base = lds_addr_u32(s);
  const char* gb = (const char*)g;
#pragma unroll
  for (int it = 0; it < 16; ++it) {
    unsigned off = (unsigned)(tid + it * 128) * 16u;   // bytes
    asm volatile("global_load_async_to_lds_b128 %0, %1, off"
                 :: "v"(base + off), "v"(gb + off)
                 : "memory");
  }
  asm volatile("s_wait_asynccnt 0x0" ::: "memory");
  __syncthreads();
}

// ---- A-fragment: 16x32 bf16, per ISA layout (lanes 0-15: M=0-15;
// VGPR0-3 hold K = hi*8 + 0..7, VGPR4-7 hold K = 16 + hi*8 + 0..7) ----
__device__ __forceinline__ v16bf load_a_frag_f32(const float* base, int lane, int kchunk) {
  int m = lane & 15, hi = lane >> 4;
  const float* p0 = base + m * NH + kchunk * 32 + hi * 8;
  const float4* q0 = (const float4*)p0;
  const float4* q1 = (const float4*)(p0 + 16);
  float4 a0 = q0[0], a1 = q0[1], a2 = q1[0], a3 = q1[1];
  v16bf a;
  a[0]=(__bf16)a0.x; a[1]=(__bf16)a0.y; a[2]=(__bf16)a0.z; a[3]=(__bf16)a0.w;
  a[4]=(__bf16)a1.x; a[5]=(__bf16)a1.y; a[6]=(__bf16)a1.z; a[7]=(__bf16)a1.w;
  a[8]=(__bf16)a2.x; a[9]=(__bf16)a2.y; a[10]=(__bf16)a2.z; a[11]=(__bf16)a2.w;
  a[12]=(__bf16)a3.x; a[13]=(__bf16)a3.y; a[14]=(__bf16)a3.z; a[15]=(__bf16)a3.w;
  return a;
}

// ---- Repack weights (o,i layout, out_o = sum_i z_i*W[o,i]) into per-lane
// contiguous B-fragments: frag[((nt*4+kc)*32+lane)*16 + t] = W[o][k],
// o = nt*16 + (lane&15), k = kc*32 + (lane>>4)*8 + (t&7) + (t>>3)*16  (dense-B 32x16 layout)
__global__ void prep_weights(const float* __restrict__ Uw, const float* __restrict__ Vw,
                             const float* __restrict__ Aw, const float* __restrict__ Bw,
                             const float* __restrict__ Cw, __bf16* __restrict__ frags) {
  int w = blockIdx.y;
  const float* W = (w==0)?Uw:(w==1)?Vw:(w==2)?Aw:(w==3)?Bw:Cw;
  int idx = blockIdx.x * 256 + threadIdx.x;          // 0..16383
  int t = idx & 15, lane = (idx >> 4) & 31, kc = (idx >> 9) & 3, nt = idx >> 11;
  int o = nt * 16 + (lane & 15);
  int k = kc * 32 + (lane >> 4) * 8 + (t & 7) + (t >> 3) * 16;
  frags[w * 16384 + idx] = (__bf16)W[o * NH + k];
}

// ---- Uh/Vh/Ah/Bh : 1024x128 rows, 4 GEMMs via WMMA ----
__global__ __launch_bounds__(128)
void node_linears(const float* __restrict__ h, const __bf16* __restrict__ frags,
                  const float* __restrict__ Ub, const float* __restrict__ Vb,
                  const float* __restrict__ Ab, const float* __restrict__ Bb,
                  float* __restrict__ Uh, float* __restrict__ Vh,
                  float* __restrict__ Ah, float* __restrict__ Bh) {
  int tid = threadIdx.x, lane = tid & 31, wave = tid >> 5;
  int tile = blockIdx.x * 4 + wave;                  // 0..63
  int row0 = tile * 16;
  int m = lane & 15, mb = (lane >> 4) * 8;
  v16bf afr[4];
#pragma unroll
  for (int k = 0; k < 4; ++k) afr[k] = load_a_frag_f32(h + row0 * NH, lane, k);
#pragma unroll
  for (int w = 0; w < 4; ++w) {
    const __bf16* wf = frags + w * 16384;
    const float* bias = (w==0)?Ub:(w==1)?Vb:(w==2)?Ab:Bb;
    float* outp = (w==0)?Uh:(w==1)?Vh:(w==2)?Ah:Bh;
#pragma unroll
    for (int nt = 0; nt < 8; ++nt) {
      v16bf bfr[4];
#pragma unroll
      for (int k = 0; k < 4; ++k)
        bfr[k] = *(const v16bf*)(wf + ((nt * 4 + k) * 32 + lane) * 16);
      v8f acc = {};
#pragma unroll
      for (int k = 0; k < 4; ++k) acc = WMMA_BF16(afr[k], bfr[k], acc);
      int n = nt * 16 + m;
      float bv = bias[n];
#pragma unroll
      for (int r = 0; r < 8; ++r) outp[(row0 + mb + r) * NH + n] = acc[r] + bv;
    }
  }
}

// ---- Pass 1: per-(b,i) workgroup.  GEMM Ce tiles + gates + aggregation +
// pa/x-update + BN partial stats.  No global atomics. ----
__global__ __launch_bounds__(128)
void e_pass1(const float* __restrict__ e, const float* __restrict__ x,
             const int* __restrict__ graph,
             const float* __restrict__ Ahf, const float* __restrict__ Bhf,
             const float* __restrict__ Vhf,
             const __bf16* __restrict__ cwf, const float* __restrict__ Cb,
             const float* __restrict__ dw, const float* __restrict__ db,
             const float* __restrict__ ew, const float* __restrict__ eb,
             float* __restrict__ hagg, float* __restrict__ xupd,
             float* __restrict__ stats_partial) {
  __shared__ __align__(32) __bf16 shfrag[16384];     // 32 KB Cw fragments
  __shared__ float l_hagg[4][128];
  __shared__ float l_stats[4][256];
  __shared__ float l_x[4][2];
  int tid = threadIdx.x, lane = tid & 31, wave = tid >> 5;

  stage_frags_async(cwf, shfrag, tid);               // async copy + barrier

  int bi = blockIdx.x;                               // (b,i)
  int b = bi >> 8, i = bi & 255;
  int m = lane & 15, hi = lane >> 4, mb = hi * 8;

  float xi0 = x[(b * NV + i) * 2 + 0], xi1 = x[(b * NV + i) * 2 + 1];

  float Bh_n[8], cb_n[8], dw_n[8], db_n[8], ew_n[8];
#pragma unroll
  for (int nt = 0; nt < 8; ++nt) {
    int n = nt * 16 + m;
    Bh_n[nt] = Bhf[(b * NV + i) * NH + n];
    cb_n[nt] = Cb[n]; dw_n[nt] = dw[n]; db_n[nt] = db[n]; ew_n[nt] = ew[n];
  }
  float ebv = eb[0];

  float haggreg[8] = {0,0,0,0,0,0,0,0};
  float ssum[8]    = {0,0,0,0,0,0,0,0};
  float ssq[8]     = {0,0,0,0,0,0,0,0};
  float xacc0 = 0.f, xacc1 = 0.f;

  for (int tt = 0; tt < 4; ++tt) {
    int j0 = (wave * 4 + tt) * 16;
    int rowbase = (b * NV + i) * NV + j0;
    const float* abase = e + (long)rowbase * NH;
    __builtin_prefetch(abase + 16 * NH, 0, 1);       // next j-tile
    v16bf afr[4];
#pragma unroll
    for (int k = 0; k < 4; ++k) afr[k] = load_a_frag_f32(abase, lane, k);

    float dist_r[8], dx0[8], dx1[8], mk[8];
#pragma unroll
    for (int r = 0; r < 8; ++r) {
      int j = j0 + mb + r;
      float xj0 = x[(b * NV + j) * 2 + 0], xj1 = x[(b * NV + j) * 2 + 1];
      float d0 = xi0 - xj0, d1 = xi1 - xj1;
      dx0[r] = d0; dx1[r] = d1;
      float d2 = d0 * d0 + d1 * d1;
      dist_r[r] = d2 > 0.f ? sqrtf(d2) : 0.f;
      mk[r] = (graph[(b * NV + i) * NV + j] == 1) ? 0.f : 1.f;
    }

    float paacc[8] = {0,0,0,0,0,0,0,0};
#pragma unroll
    for (int nt = 0; nt < 8; ++nt) {
      v16bf bfr[4];
#pragma unroll
      for (int k = 0; k < 4; ++k)
        bfr[k] = *(const v16bf*)(shfrag + ((nt * 4 + k) * 32 + lane) * 16);
      v8f acc = {};
#pragma unroll
      for (int k = 0; k < 4; ++k) acc = WMMA_BF16(afr[k], bfr[k], acc);
      int n = nt * 16 + m;
#pragma unroll
      for (int r = 0; r < 8; ++r) {
        int j = j0 + mb + r;
        float en = acc[r] + Ahf[(b * NV + j) * NH + n] + Bh_n[nt]
                 + cb_n[nt] + dist_r[r] * dw_n[nt] + db_n[nt];
        ssum[nt] += en;
        ssq[nt]  += en * en;
        float g = 1.f / (1.f + __expf(-en));
        haggreg[nt] += mk[r] * g * Vhf[(b * NV + j) * NH + n];
        paacc[r] += en * ew_n[nt];
      }
    }
    // pa: reduce the 128-col dot across the 16 lanes of each half-wave
#pragma unroll
    for (int r = 0; r < 8; ++r) {
      float p = paacc[r];
      p += __shfl_xor(p, 1, 16);
      p += __shfl_xor(p, 2, 16);
      p += __shfl_xor(p, 4, 16);
      p += __shfl_xor(p, 8, 16);
      float pa = 1.f / (1.f + __expf(-(p + ebv)));
      if (m == 0) { xacc0 += pa * dx0[r]; xacc1 += pa * dx1[r]; }
    }
  }

  // fold the two half-waves (same channel n, different rows) and stage to LDS
#pragma unroll
  for (int nt = 0; nt < 8; ++nt) {
    int n = nt * 16 + m;
    float v = haggreg[nt] + __shfl_xor(haggreg[nt], 16, 32);
    float s = ssum[nt]    + __shfl_xor(ssum[nt], 16, 32);
    float q = ssq[nt]     + __shfl_xor(ssq[nt], 16, 32);
    if (hi == 0) { l_hagg[wave][n] = v; l_stats[wave][n] = s; l_stats[wave][128 + n] = q; }
  }
  float x0 = xacc0 + __shfl_xor(xacc0, 16, 32);
  float x1 = xacc1 + __shfl_xor(xacc1, 16, 32);
  if (lane == 0) { l_x[wave][0] = x0; l_x[wave][1] = x1; }
  __syncthreads();

  if (tid < 128) {
    float s = l_hagg[0][tid] + l_hagg[1][tid] + l_hagg[2][tid] + l_hagg[3][tid];
    hagg[bi * NH + tid] = s;
  }
  for (int s = tid; s < 256; s += 128) {
    stats_partial[bi * 256 + s] =
        l_stats[0][s] + l_stats[1][s] + l_stats[2][s] + l_stats[3][s];
  }
  if (tid < 2) xupd[bi * 2 + tid] = l_x[0][tid] + l_x[1][tid] + l_x[2][tid] + l_x[3][tid];
}

// ---- reduce BN stats for e, produce per-channel scale/shift ----
__global__ void reduce_estats(const float* __restrict__ partial,
                              const float* __restrict__ gamma_e,
                              const float* __restrict__ beta_e,
                              float* __restrict__ scale_e, float* __restrict__ shift_e) {
  int n = threadIdx.x;                               // 0..127
  float s = 0.f, q = 0.f;
  for (int w = 0; w < NB * NV; ++w) {
    s += partial[w * 256 + n];
    q += partial[w * 256 + 128 + n];
  }
  const float cnt = (float)M_E;
  float mu = s / cnt;
  float var = q / cnt - mu * mu;
  float sc = gamma_e[n] * rsqrtf(var + EPS);
  scale_e[n] = sc;
  shift_e[n] = beta_e[n] - mu * sc;
}

// ---- h: BN over 1024 rows, relu, residual ----
__global__ __launch_bounds__(256)
void h_finalize(const float* __restrict__ h, const float* __restrict__ Uh,
                const float* __restrict__ hagg,
                const float* __restrict__ gamma_h, const float* __restrict__ beta_h,
                float* __restrict__ out_h) {
  __shared__ float red[256], red2[256];
  int n = blockIdx.x, t = threadIdx.x;
  float v[4]; float s = 0.f, q = 0.f;
#pragma unroll
  for (int r = 0; r < 4; ++r) {
    int row = t + r * 256;
    float z = Uh[row * NH + n] + hagg[row * NH + n];
    v[r] = z; s += z; q += z * z;
  }
  red[t] = s; red2[t] = q; __syncthreads();
  for (int off = 128; off; off >>= 1) {
    if (t < off) { red[t] += red[t + off]; red2[t] += red2[t + off]; }
    __syncthreads();
  }
  float mu = red[0] / 1024.f, var = red2[0] / 1024.f - mu * mu;
  float sc = gamma_h[n] * rsqrtf(var + EPS), sh = beta_h[n] - mu * sc;
#pragma unroll
  for (int r = 0; r < 4; ++r) {
    int row = t + r * 256;
    float z = v[r] * sc + sh;
    z = z > 0.f ? z : 0.f;
    out_h[row * NH + n] = h[row * NH + n] + z;
  }
}

__global__ void x_finalize(const float* __restrict__ x, const float* __restrict__ xupd,
                           const float* __restrict__ c, float* __restrict__ out_x) {
  float cv = c[0];
  for (int p = threadIdx.x; p < NB * NV * 2; p += 256) out_x[p] = x[p] + cv * xupd[p];
}

// ---- Pass 2: recompute e_new tiles, BN+relu into LDS, coalesced residual ----
__global__ __launch_bounds__(128)
void e_pass2(const float* __restrict__ e, const float* __restrict__ x,
             const float* __restrict__ Ahf, const float* __restrict__ Bhf,
             const __bf16* __restrict__ cwf, const float* __restrict__ Cb,
             const float* __restrict__ dw, const float* __restrict__ db,
             const float* __restrict__ scale_e, const float* __restrict__ shift_e,
             float* __restrict__ out_e) {
  __shared__ __align__(32) __bf16 shfrag[16384];     // 32 KB Cw fragments
  __shared__ __align__(16) float tilebuf[64 * 128];  // 32 KB
  int tid = threadIdx.x, lane = tid & 31, wave = tid >> 5;

  stage_frags_async(cwf, shfrag, tid);               // async copy + barrier

  int tile = blockIdx.x * 4 + wave;                  // 0..16383
  int row0 = tile * 16;
  int b = row0 >> 16, i = (row0 >> 8) & 255, j0 = row0 & 255;
  int m = lane & 15, hi = lane >> 4, mb = hi * 8;

  const float* abase = e + (long)row0 * NH;
  v16bf afr[4];
#pragma unroll
  for (int k = 0; k < 4; ++k) afr[k] = load_a_frag_f32(abase, lane, k);

  float xi0 = x[(b * NV + i) * 2 + 0], xi1 = x[(b * NV + i) * 2 + 1];
  float dist_r[8];
#pragma unroll
  for (int r = 0; r < 8; ++r) {
    int j = j0 + mb + r;
    float d0 = xi0 - x[(b * NV + j) * 2 + 0];
    float d1 = xi1 - x[(b * NV + j) * 2 + 1];
    float d2 = d0 * d0 + d1 * d1;
    dist_r[r] = d2 > 0.f ? sqrtf(d2) : 0.f;
  }

#pragma unroll
  for (int nt = 0; nt < 8; ++nt) {
    v16bf bfr[4];
#pragma unroll
    for (int k = 0; k < 4; ++k)
      bfr[k] = *(const v16bf*)(shfrag + ((nt * 4 + k) * 32 + lane) * 16);
    v8f acc = {};
#pragma unroll
    for (int k = 0; k < 4; ++k) acc = WMMA_BF16(afr[k], bfr[k], acc);
    int n = nt * 16 + m;
    float bh = Bhf[(b * NV + i) * NH + n];
    float cb = Cb[n], dwn = dw[n], dbn = db[n];
    float sc = scale_e[n], sh = shift_e[n];
#pragma unroll
    for (int r = 0; r < 8; ++r) {
      int j = j0 + mb + r;
      float en = acc[r] + Ahf[(b * NV + j) * NH + n] + bh + cb + dist_r[r] * dwn + dbn;
      float z = en * sc + sh;
      z = z > 0.f ? z : 0.f;
      tilebuf[wave * 2048 + (mb + r) * NH + n] = z;
    }
  }
  __syncthreads();

  // coalesced residual: 64 rows * 128 cols per WG
  const float4* esrc = (const float4*)(e + (long)blockIdx.x * 8192);
  const float4* lsrc = (const float4*)tilebuf;
  float4* dst = (float4*)(out_e + (long)blockIdx.x * 8192);
#pragma unroll
  for (int it = 0; it < 16; ++it) {
    int idx = it * 128 + tid;
    float4 ev = esrc[idx], lv = lsrc[idx];
    float4 o;
    o.x = ev.x + lv.x; o.y = ev.y + lv.y; o.z = ev.z + lv.z; o.w = ev.w + lv.w;
    dst[idx] = o;
  }
}

// ---- workspace layout (bytes) ----
#define OFF_FRAGS   ((size_t)0)                      // 5 * 16384 bf16 = 160 KB
#define OFF_UH      ((size_t)163840)
#define OFF_VH      (OFF_UH + (size_t)NB*NV*NH*4)
#define OFF_AH      (OFF_VH + (size_t)NB*NV*NH*4)
#define OFF_BH      (OFF_AH + (size_t)NB*NV*NH*4)
#define OFF_HAGG    (OFF_BH + (size_t)NB*NV*NH*4)
#define OFF_XUPD    (OFF_HAGG + (size_t)NB*NV*NH*4)
#define OFF_PART    (OFF_XUPD + (size_t)NB*NV*2*4)
#define OFF_SCALE   (OFF_PART + (size_t)NB*NV*256*4)
#define OFF_SHIFT   (OFF_SCALE + (size_t)512)

extern "C" void kernel_launch(void* const* d_in, const int* in_sizes, int n_in,
                              void* d_out, int out_size, void* d_ws, size_t ws_size,
                              hipStream_t stream) {
  const float* h     = (const float*)d_in[0];
  const float* e     = (const float*)d_in[1];
  const float* x     = (const float*)d_in[2];
  const int*   graph = (const int*)  d_in[3];
  const float* Uw = (const float*)d_in[4],  *Ub = (const float*)d_in[5];
  const float* Vw = (const float*)d_in[6],  *Vb = (const float*)d_in[7];
  const float* Aw = (const float*)d_in[8],  *Ab = (const float*)d_in[9];
  const float* Bw = (const float*)d_in[10], *Bb = (const float*)d_in[11];
  const float* Cw = (const float*)d_in[12], *Cb = (const float*)d_in[13];
  const float* dw = (const float*)d_in[14], *db = (const float*)d_in[15];
  const float* ew = (const float*)d_in[16], *eb = (const float*)d_in[17];
  const float* c  = (const float*)d_in[18];
  const float* gamma_h = (const float*)d_in[19], *beta_h = (const float*)d_in[20];
  const float* gamma_e = (const float*)d_in[21], *beta_e = (const float*)d_in[22];

  char* ws = (char*)d_ws;
  __bf16* frags   = (__bf16*)(ws + OFF_FRAGS);
  float* Uh       = (float*)(ws + OFF_UH);
  float* Vh       = (float*)(ws + OFF_VH);
  float* Ah       = (float*)(ws + OFF_AH);
  float* Bh       = (float*)(ws + OFF_BH);
  float* hagg     = (float*)(ws + OFF_HAGG);
  float* xupd     = (float*)(ws + OFF_XUPD);
  float* part     = (float*)(ws + OFF_PART);
  float* scale_e  = (float*)(ws + OFF_SCALE);
  float* shift_e  = (float*)(ws + OFF_SHIFT);

  float* out_h = (float*)d_out;
  float* out_e = out_h + (size_t)NB * NV * NH;
  float* out_x = out_e + (size_t)NB * NV * NV * NH;

  prep_weights<<<dim3(64, 5), 256, 0, stream>>>(Uw, Vw, Aw, Bw, Cw, frags);
  node_linears<<<16, 128, 0, stream>>>(h, frags, Ub, Vb, Ab, Bb, Uh, Vh, Ah, Bh);
  e_pass1<<<NB * NV, 128, 0, stream>>>(e, x, graph, Ah, Bh, Vh,
                                       frags + 4 * 16384, Cb, dw, db, ew, eb,
                                       hagg, xupd, part);
  reduce_estats<<<1, 128, 0, stream>>>(part, gamma_e, beta_e, scale_e, shift_e);
  h_finalize<<<NH, 256, 0, stream>>>(h, Uh, hagg, gamma_h, beta_h, out_h);
  x_finalize<<<1, 256, 0, stream>>>(x, xupd, c, out_x);
  e_pass2<<<M_E / 64, 128, 0, stream>>>(e, x, Ah, Bh, frags + 4 * 16384, Cb, dw, db,
                                        scale_e, shift_e, out_e);
  (void)in_sizes; (void)n_in; (void)out_size; (void)ws_size;
}
